// AngularPenaltySMLoss_13786845020669
// MI455X (gfx1250) — compile-verified, matrix operand order checked by reading, not run on hardware
//
#include <hip/hip_runtime.h>
#include <hip/hip_fp16.h>

typedef __attribute__((ext_vector_type(16))) _Float16 v16h;
typedef __attribute__((ext_vector_type(4)))  _Float16 v4h;
typedef __attribute__((ext_vector_type(8)))  float    v8f;

#define NB 2048      // batch
#define NC 32768     // classes
#define ND 512       // feature dim
#define BM 128       // block tile M
#define BN 128       // block tile N
#define BK 64        // block tile K (two WMMA K-steps per stage)
#define KSTAGES (ND / BK)   // 8

__constant__ float S_SCALE = 64.0f;
__constant__ float MARGIN  = 0.5f;
__constant__ float ACLIP   = 1e-7f;

// ---------------------------------------------------------------------------
__global__ void init_kernel(float* __restrict__ rowsum, float* __restrict__ out) {
    int i = blockIdx.x * blockDim.x + threadIdx.x;
    if (i < NB) rowsum[i] = 0.0f;
    if (i == 0) *out = 0.0f;
}

// ---------------------------------------------------------------------------
// Per-row reciprocal L2 norm: rinv[row] = 1 / max(||x_row||, 1e-12)
// ---------------------------------------------------------------------------
__global__ __launch_bounds__(128) void rownorm_kernel(const float* __restrict__ x,
                                                      float* __restrict__ rinv) {
    const int row = blockIdx.x;
    const float4* xr = (const float4*)(x + (size_t)row * ND);
    float4 v = xr[threadIdx.x];                 // 128 threads * 4 = 512
    float ss = v.x * v.x + v.y * v.y + v.z * v.z + v.w * v.w;
    for (int off = 16; off; off >>= 1) ss += __shfl_xor(ss, off, 32);

    __shared__ float smem[4];
    const int lane = threadIdx.x & 31, wid = threadIdx.x >> 5;
    if (lane == 0) smem[wid] = ss;
    __syncthreads();
    if (threadIdx.x == 0) {
        float t = smem[0] + smem[1] + smem[2] + smem[3];
        rinv[row] = 1.0f / fmaxf(sqrtf(t), 1e-12f);
    }
}

// ---------------------------------------------------------------------------
// Fused normalized-GEMM + exp-rowsum + target extraction.
// Ping-pong LDS (64 KB), software-pipelined global->reg->LDS staging with
// f32->f16 conversion fused in; 16 WMMAs per stage, one barrier per stage.
// ---------------------------------------------------------------------------
__global__ __launch_bounds__(256) void arcface_gemm_kernel(
    const float* __restrict__ feat,    // [NB, ND] f32
    const float* __restrict__ weight,  // [NC, ND] f32
    const int*   __restrict__ ytrue,   // [NB]
    const float* __restrict__ frinv,   // [NB]
    const float* __restrict__ wrinv,   // [NC]
    float* __restrict__ rowsum,        // [NB]
    float* __restrict__ tgt)           // [NB]
{
    __shared__ alignas(32) _Float16 Ah[2][BM * BK];   // 2 x 16 KB
    __shared__ alignas(32) _Float16 Bh[2][BN * BK];   // 2 x 16 KB

    const int bm   = blockIdx.y * BM;
    const int bn   = blockIdx.x * BN;
    const int tid  = threadIdx.x;
    const int lane = tid & 31;
    const int wave = tid >> 5;
    const int half = lane >> 4;        // 0: lanes 0-15, 1: lanes 16-31
    const int l16  = lane & 15;

    // staging geometry: thread stages rows (tid>>4)+16*it, cols [(tid&15)*4, +4)
    const int srow0 = tid >> 4;        // 0..15
    const int scol  = (tid & 15) * 4;  // 0..60

    // row scales are k-invariant: hoist out of the k-loop
    float fs[8], wsc[8];
    #pragma unroll
    for (int it = 0; it < 8; ++it) {
        fs[it]  = frinv[bm + srow0 + it * 16];
        wsc[it] = wrinv[bn + srow0 + it * 16];
    }

    v8f acc[8] = {};
    float4 areg[8], breg[8];

    // ---- prologue: prefetch stage 0 and fill buffer 0 ----
    #pragma unroll
    for (int it = 0; it < 8; ++it) {
        const int r = srow0 + it * 16;
        areg[it] = *(const float4*)(feat   + (size_t)(bm + r) * ND + scol);
        breg[it] = *(const float4*)(weight + (size_t)(bn + r) * ND + scol);
    }
    #pragma unroll
    for (int it = 0; it < 8; ++it) {
        const int r = srow0 + it * 16;
        v4h ah, bh;
        ah.x = (_Float16)(areg[it].x * fs[it]);
        ah.y = (_Float16)(areg[it].y * fs[it]);
        ah.z = (_Float16)(areg[it].z * fs[it]);
        ah.w = (_Float16)(areg[it].w * fs[it]);
        *(v4h*)&Ah[0][r * BK + scol] = ah;
        bh.x = (_Float16)(breg[it].x * wsc[it]);
        bh.y = (_Float16)(breg[it].y * wsc[it]);
        bh.z = (_Float16)(breg[it].z * wsc[it]);
        bh.w = (_Float16)(breg[it].w * wsc[it]);
        *(v4h*)&Bh[0][r * BK + scol] = bh;
    }
    __syncthreads();

    // ---- main pipelined loop ----
    for (int ks = 0; ks < KSTAGES; ++ks) {
        const int buf = ks & 1;

        // issue next stage's global loads first (latency hides under WMMAs)
        if (ks + 1 < KSTAGES) {
            const int kn = (ks + 1) * BK;
            #pragma unroll
            for (int it = 0; it < 8; ++it) {
                const int r = srow0 + it * 16;
                areg[it] = *(const float4*)(feat   + (size_t)(bm + r) * ND + kn + scol);
                breg[it] = *(const float4*)(weight + (size_t)(bn + r) * ND + kn + scol);
            }
        }

        // 16 WMMAs from current buffer; A fragments reused across all 8 tiles
        const int arow = wave * 16 + l16;
        const v16h a0 = *(const v16h*)&Ah[buf][arow * BK +      half * 16];
        const v16h a1 = *(const v16h*)&Ah[buf][arow * BK + 32 + half * 16];
        #pragma unroll
        for (int t = 0; t < 8; ++t) {
            const int brow = t * 16 + l16;
            const v16h b0 = *(const v16h*)&Bh[buf][brow * BK +      half * 16];
            const v16h b1 = *(const v16h*)&Bh[buf][brow * BK + 32 + half * 16];
            acc[t] = __builtin_amdgcn_wmma_f32_16x16x32_f16(
                false, a0, false, b0, (short)0, acc[t], false, false);
            acc[t] = __builtin_amdgcn_wmma_f32_16x16x32_f16(
                false, a1, false, b1, (short)0, acc[t], false, false);
        }

        // convert + store next stage into the other buffer, then one barrier
        if (ks + 1 < KSTAGES) {
            #pragma unroll
            for (int it = 0; it < 8; ++it) {
                const int r = srow0 + it * 16;
                v4h ah, bh;
                ah.x = (_Float16)(areg[it].x * fs[it]);
                ah.y = (_Float16)(areg[it].y * fs[it]);
                ah.z = (_Float16)(areg[it].z * fs[it]);
                ah.w = (_Float16)(areg[it].w * fs[it]);
                *(v4h*)&Ah[buf ^ 1][r * BK + scol] = ah;
                bh.x = (_Float16)(breg[it].x * wsc[it]);
                bh.y = (_Float16)(breg[it].y * wsc[it]);
                bh.z = (_Float16)(breg[it].z * wsc[it]);
                bh.w = (_Float16)(breg[it].w * wsc[it]);
                *(v4h*)&Bh[buf ^ 1][r * BK + scol] = bh;
            }
            __syncthreads();
        }
    }

    // ---- epilogue: exp row-sums + target logit extraction ----
    const int rbase = bm + wave * 16 + half * 8;
    int yrow[8];
    #pragma unroll
    for (int j = 0; j < 8; ++j) yrow[j] = ytrue[rbase + j];

    float partial[8] = {};
    #pragma unroll
    for (int t = 0; t < 8; ++t) {
        const int col = bn + t * 16 + l16;
        #pragma unroll
        for (int j = 0; j < 8; ++j) {
            const float v = acc[t][j];
            partial[j] += __expf(S_SCALE * v);
            if (yrow[j] == col) tgt[rbase + j] = v;   // unique writer per row
        }
    }

    #pragma unroll
    for (int j = 0; j < 8; ++j) {
        float p = partial[j];
        p += __shfl_xor(p, 1, 32);
        p += __shfl_xor(p, 2, 32);
        p += __shfl_xor(p, 4, 32);
        p += __shfl_xor(p, 8, 32);
        if (l16 == 0) atomicAdd(&rowsum[rbase + j], p);
    }
}

// ---------------------------------------------------------------------------
__global__ void finalize_kernel(const float* __restrict__ rowsum,
                                const float* __restrict__ tgt,
                                float* __restrict__ out) {
    const int i = blockIdx.x * blockDim.x + threadIdx.x;
    if (i >= NB) return;
    const float t   = tgt[i];
    const float tc  = fminf(fmaxf(t, -1.0f + ACLIP), 1.0f - ACLIP);
    const float num = S_SCALE * cosf(acosf(tc) + MARGIN);
    const float excl  = rowsum[i] - __expf(S_SCALE * t);
    const float denom = __expf(num) + excl;
    const float L = num - __logf(denom);

    __shared__ float sred[8];
    float p = -L / (float)NB;
    for (int off = 16; off; off >>= 1) p += __shfl_xor(p, off, 32);
    const int lane = threadIdx.x & 31, wid = threadIdx.x >> 5;
    if (lane == 0) sred[wid] = p;
    __syncthreads();
    if (threadIdx.x == 0) {
        float s = 0.0f;
        for (int w = 0; w < (int)(blockDim.x >> 5); ++w) s += sred[w];
        atomicAdd(out, s);
    }
}

// ---------------------------------------------------------------------------
extern "C" void kernel_launch(void* const* d_in, const int* in_sizes, int n_in,
                              void* d_out, int out_size, void* d_ws, size_t ws_size,
                              hipStream_t stream) {
    const float* feat   = (const float*)d_in[0];   // [2048, 512]
    const int*   ytrue  = (const int*)d_in[1];     // [2048]
    const float* weight = (const float*)d_in[2];   // [32768, 512]
    float*       out    = (float*)d_out;           // scalar

    float* frinv  = (float*)d_ws;                  // [NB]
    float* wrinv  = frinv + NB;                    // [NC]
    float* rowsum = wrinv + NC;                    // [NB]
    float* tgt    = rowsum + NB;                   // [NB]

    init_kernel<<<(NB + 255) / 256, 256, 0, stream>>>(rowsum, out);
    rownorm_kernel<<<NB, 128, 0, stream>>>(feat, frinv);
    rownorm_kernel<<<NC, 128, 0, stream>>>(weight, wrinv);

    dim3 grid(NC / BN, NB / BM);   // (256, 16)
    arcface_gemm_kernel<<<grid, 256, 0, stream>>>(feat, weight, ytrue,
                                                  frinv, wrinv, rowsum, tgt);

    finalize_kernel<<<(NB + 255) / 256, 256, 0, stream>>>(rowsum, tgt, out);
}
